// Seq2SeqPRI_45749991637025
// MI455X (gfx1250) — matrix operand order repaired and checked by reading, not compiled, not run on hardware
//
#include <hip/hip_runtime.h>

// ---------------- CDNA5 wave32 WMMA bf16 infrastructure ----------------
typedef __attribute__((ext_vector_type(16))) __bf16 v16bf;
typedef __attribute__((ext_vector_type(8)))  __bf16 v8bf;
typedef __attribute__((ext_vector_type(8)))  float  v8f;

__device__ __forceinline__ __bf16 f2bf(float x) {
  union { float f; unsigned u; } v; v.f = x;
  unsigned r = v.u + 0x7fffu + ((v.u >> 16) & 1u);   // round-to-nearest-even
  unsigned short h = (unsigned short)(r >> 16);
  __bf16 o; __builtin_memcpy(&o, &h, 2); return o;
}
__device__ __forceinline__ float bf2f(__bf16 x) {
  unsigned short h; __builtin_memcpy(&h, &x, 2);
  union { unsigned u; float f; } v; v.u = ((unsigned)h) << 16; return v.f;
}
__device__ __forceinline__ float sigmoidf(float x) { return 1.0f / (1.0f + __expf(-x)); }

// A-matrix 16x32 bf16 fragment (ISA 7.12.2): lane m=lane&15, hi=lane>>4;
// elements e<8 at K=e+8*hi, e>=8 at K=e+8+8*hi  -> two contiguous b128 loads.
__device__ __forceinline__ v16bf load_fragA(const __bf16* __restrict__ A, int lda,
                                            int m0, int k, int lane) {
  int m = m0 + (lane & 15), hi = lane >> 4;
  const __bf16* p = A + (size_t)m * lda + k + 8 * hi;
  v8bf lo = *(const v8bf*)p;
  v8bf hh = *(const v8bf*)(p + 16);
  v16bf r;
#pragma unroll
  for (int i = 0; i < 8; ++i) { r[i] = lo[i]; r[i + 8] = hh[i]; }
  return r;
}
// B-matrix 32x16 bf16 fragment: lane n=lane&15; e<8 at K=e+8*hi, e>=8 at K=e+8+8*hi
// reading W stored row-major as [N][K] (i.e. the original weight layout).
__device__ __forceinline__ v16bf load_fragB(const __bf16* __restrict__ W, int ldw,
                                            int n0, int k, int lane) {
  int n = n0 + (lane & 15), hi = lane >> 4;
  const __bf16* p = W + (size_t)n * ldw + k + 8 * hi;
  v8bf lo = *(const v8bf*)p;
  v8bf hh = *(const v8bf*)(p + 16);
  v16bf r;
#pragma unroll
  for (int i = 0; i < 8; ++i) { r[i] = lo[i]; r[i + 8] = hh[i]; }
  return r;
}
#define WMMA_BF16(acc, a, b) \
  acc = __builtin_amdgcn_wmma_f32_16x16x32_bf16(false, (a), false, (b), (short)0, (acc), false, false)

// ---------------- generic GEMM: C[M,N] = A[M,K](bf16) * W[N,K]^T(bf16) (+bias) ----------------
// block = 256 threads = 8 waves; wave tile 16x64; block tile 32x256.
// grid = (N/256, M/32). M,N,K must be multiples of 32/256/32 (true for all uses).
__global__ void gemm_bf16_kernel(const __bf16* __restrict__ A, int lda,
                                 const __bf16* __restrict__ W, int ldw,
                                 float* __restrict__ C, int ldc,
                                 int K, const float* __restrict__ bias) {
  int lane = threadIdx.x & 31, wave = threadIdx.x >> 5;
  int m0 = blockIdx.y * 32 + (wave & 1) * 16;
  int n0 = blockIdx.x * 256 + (wave >> 1) * 64;
  v8f a0 = {}, a1 = {}, a2 = {}, a3 = {};
  for (int k = 0; k < K; k += 32) {
    v16bf af = load_fragA(A, lda, m0, k, lane);
    v16bf b0 = load_fragB(W, ldw, n0, k, lane);
    v16bf b1 = load_fragB(W, ldw, n0 + 16, k, lane);
    v16bf b2 = load_fragB(W, ldw, n0 + 32, k, lane);
    v16bf b3 = load_fragB(W, ldw, n0 + 48, k, lane);
    WMMA_BF16(a0, af, b0); WMMA_BF16(a1, af, b1);
    WMMA_BF16(a2, af, b2); WMMA_BF16(a3, af, b3);
  }
  int cn = n0 + (lane & 15), hi = lane >> 4;
  float bb0 = bias ? bias[cn] : 0.f;
  float bb1 = bias ? bias[cn + 16] : 0.f;
  float bb2 = bias ? bias[cn + 32] : 0.f;
  float bb3 = bias ? bias[cn + 48] : 0.f;
#pragma unroll
  for (int r = 0; r < 8; ++r) {
    size_t row = (size_t)(m0 + r + 8 * hi) * ldc;
    C[row + cn]      = a0[r] + bb0;
    C[row + cn + 16] = a1[r] + bb1;
    C[row + cn + 32] = a2[r] + bb2;
    C[row + cn + 48] = a3[r] + bb3;
  }
}

// ---------------- encoder recurrent step GEMM (both directions in one launch) -------------
// gates[dir][b][n] = [emb_tf | h_dir] @ Wcat_dir^T + bias_dir ; K = 256 + 512 = 768.
// grid = (8, 2, 2): x -> N/256, y -> M/32, z -> direction.
__global__ void enc_step_gemm_kernel(const __bf16* __restrict__ emb,   // [S][B][E]
                                     const __bf16* __restrict__ hbf,   // [2][B][H]
                                     const __bf16* __restrict__ Wf,    // [2048][768]
                                     const __bf16* __restrict__ Wb,
                                     const float* __restrict__ biasf,
                                     const float* __restrict__ biasb,
                                     float* __restrict__ gates,        // [2][B][2048]
                                     int t) {
  int dir = blockIdx.z;
  int tf = dir ? (511 - t) : t;
  const __bf16* W = dir ? Wb : Wf;
  const float* bias = dir ? biasb : biasf;
  const __bf16* hsrc = hbf + (size_t)dir * 64 * 512;
  const __bf16* xsrc = emb + (size_t)tf * 64 * 256;
  float* C = gates + (size_t)dir * 64 * 2048;
  int lane = threadIdx.x & 31, wave = threadIdx.x >> 5;
  int m0 = blockIdx.y * 32 + (wave & 1) * 16;
  int n0 = blockIdx.x * 256 + (wave >> 1) * 64;
  v8f a0 = {}, a1 = {}, a2 = {}, a3 = {};
  for (int k = 0; k < 768; k += 32) {
    const __bf16* ab; int alda, ac;
    if (k < 256) { ab = xsrc; alda = 256; ac = k; }
    else         { ab = hsrc; alda = 512; ac = k - 256; }
    v16bf af = load_fragA(ab, alda, m0, ac, lane);
    v16bf b0 = load_fragB(W, 768, n0, k, lane);
    v16bf b1 = load_fragB(W, 768, n0 + 16, k, lane);
    v16bf b2 = load_fragB(W, 768, n0 + 32, k, lane);
    v16bf b3 = load_fragB(W, 768, n0 + 48, k, lane);
    WMMA_BF16(a0, af, b0); WMMA_BF16(a1, af, b1);
    WMMA_BF16(a2, af, b2); WMMA_BF16(a3, af, b3);
  }
  int cn = n0 + (lane & 15), hi = lane >> 4;
#pragma unroll
  for (int r = 0; r < 8; ++r) {
    size_t row = (size_t)(m0 + r + 8 * hi) * 2048;
    C[row + cn]      = a0[r] + bias[cn];
    C[row + cn + 16] = a1[r] + bias[cn + 16];
    C[row + cn + 32] = a2[r] + bias[cn + 32];
    C[row + cn + 48] = a3[r] + bias[cn + 48];
  }
}

// ---------------- pointwise / prep kernels ----------------
__global__ void enc_step_point_kernel(const float* __restrict__ gates,
                                      const int* __restrict__ src_mask,
                                      float* __restrict__ h, float* __restrict__ c,
                                      __bf16* __restrict__ hbf,
                                      __bf16* __restrict__ enc_out, int t) {
  int idx = blockIdx.x * 256 + threadIdx.x;      // 2*64*512 threads
  int j = idx & 511, b = (idx >> 9) & 63, dir = idx >> 15;
  int tf = dir ? (511 - t) : t;
  int m = src_mask[b * 512 + tf];
  const float* g = gates + (size_t)((dir << 6) + b) * 2048;
  float gi = g[j], gf = g[512 + j], gg = g[1024 + j], go = g[1536 + j];
  int hidx = ((dir << 6) + b) * 512 + j;
  float cp = c[hidx], hp = h[hidx];
  float cn = sigmoidf(gf) * cp + sigmoidf(gi) * tanhf(gg);
  float hn = sigmoidf(go) * tanhf(cn);
  float ho = m ? hn : hp, co = m ? cn : cp;
  h[hidx] = ho; c[hidx] = co; hbf[hidx] = f2bf(ho);
  enc_out[((size_t)b * 512 + tf) * 1024 + dir * 512 + j] = f2bf(m ? hn : 0.0f);
}

__global__ void embed_kernel(const float* __restrict__ src, const float* __restrict__ epW,
                             const float* __restrict__ epb, __bf16* __restrict__ emb) {
  size_t idx = (size_t)blockIdx.x * 256 + threadIdx.x;  // S*B*E
  int j = idx & 255; int b = (idx >> 8) & 63; int t = (int)(idx >> 14);
  emb[idx] = f2bf(src[b * 512 + t] * (1.0f / 1000.0f) * epW[j] + epb[j]);
}

__global__ void catcvt_kernel(__bf16* __restrict__ dst, const float* __restrict__ A, int KA,
                              const float* __restrict__ B, int KB, int N) {
  size_t idx = (size_t)blockIdx.x * 256 + threadIdx.x;
  int K = KA + KB;
  if (idx >= (size_t)N * K) return;
  int k = (int)(idx % K); int n = (int)(idx / K);
  float v = (k < KA) ? A[(size_t)n * KA + k] : B[(size_t)n * KB + (k - KA)];
  dst[idx] = f2bf(v);
}

__global__ void cvt_kernel(__bf16* __restrict__ dst, const float* __restrict__ src, int n) {
  int i = blockIdx.x * 256 + threadIdx.x;
  if (i < n) dst[i] = f2bf(src[i]);
}

__global__ void addvec_kernel(float* __restrict__ d, const float* __restrict__ a,
                              const float* __restrict__ b, int n) {
  int i = blockIdx.x * 256 + threadIdx.x;
  if (i < n) d[i] = a[i] + b[i];
}

__global__ void zero_kernel(unsigned* __restrict__ p, int n) {
  int i = blockIdx.x * 256 + threadIdx.x;
  if (i < n) p[i] = 0u;
}

__global__ void fc_init_kernel(const float* __restrict__ hs, const float* __restrict__ cs,
                               const float* __restrict__ Wh, const float* __restrict__ bh,
                               const float* __restrict__ Wc, const float* __restrict__ bc,
                               float* __restrict__ hdec, float* __restrict__ cdec,
                               __bf16* __restrict__ hbf) {
  int idx = blockIdx.x * 256 + threadIdx.x;   // 2*64*512
  int j = idx & 511, b = (idx >> 9) & 63, sel = idx >> 15;
  const float* src = sel ? cs : hs;
  const float* W = sel ? Wc : Wh;
  float s = (sel ? bc : bh)[j];
  const float* w0 = W + (size_t)j * 1024;
  const float* sF = src + (size_t)b * 512;          // forward final state
  const float* sB = src + (size_t)(64 + b) * 512;   // backward final state
  for (int k = 0; k < 512; ++k) s += sF[k] * w0[k];
  for (int k = 0; k < 512; ++k) s += sB[k] * w0[512 + k];
  float v = tanhf(s);
  if (sel == 0) { hdec[b * 512 + j] = v; hbf[b * 512 + j] = f2bf(v); }
  else          { cdec[b * 512 + j] = v; }
}

// scores[b][s] = valid ? sum_h tanh(hp[b][h] + eh[b][s][h]) * v[h] : -1e9
__global__ void attn_score_kernel(const float* __restrict__ hp, const float* __restrict__ eh,
                                  const float* __restrict__ av, const int* __restrict__ src_mask,
                                  float* __restrict__ scores) {
  int b = blockIdx.y;
  int lane = threadIdx.x & 31, wave = threadIdx.x >> 5;
  int s = blockIdx.x * 8 + wave;
  const float* ehp = eh + ((size_t)b * 512 + s) * 512;
  const float* hpb = hp + (size_t)b * 512;
  float sum = 0.f;
  for (int k = lane; k < 512; k += 32) sum += tanhf(hpb[k] + ehp[k]) * av[k];
#pragma unroll
  for (int o = 16; o; o >>= 1) sum += __shfl_xor(sum, o, 32);
  if (lane == 0) scores[b * 512 + s] = src_mask[b * 512 + s] ? sum : -1000000000.0f;
}

__global__ void softmax512_kernel(const float* __restrict__ scores, float* __restrict__ w) {
  int b = blockIdx.x, tid = threadIdx.x;     // 256 threads, 2 elems each
  __shared__ float sh[8];
  float x0 = scores[b * 512 + tid], x1 = scores[b * 512 + 256 + tid];
  float mx = fmaxf(x0, x1);
#pragma unroll
  for (int o = 16; o; o >>= 1) mx = fmaxf(mx, __shfl_xor(mx, o, 32));
  if ((tid & 31) == 0) sh[tid >> 5] = mx;
  __syncthreads();
  float bm = sh[0];
#pragma unroll
  for (int i = 1; i < 8; ++i) bm = fmaxf(bm, sh[i]);
  float e0 = __expf(x0 - bm), e1 = __expf(x1 - bm);
  float s = e0 + e1;
#pragma unroll
  for (int o = 16; o; o >>= 1) s += __shfl_xor(s, o, 32);
  __syncthreads();
  if ((tid & 31) == 0) sh[tid >> 5] = s;
  __syncthreads();
  float bs = 0.f;
#pragma unroll
  for (int i = 0; i < 8; ++i) bs += sh[i];
  float inv = 1.0f / bs;
  w[b * 512 + tid] = e0 * inv;
  w[b * 512 + 256 + tid] = e1 * inv;
}

// ctx[b][j] = sum_s w[b][s] * enc_out[b][s][j]   (enc_out bf16, L2-resident)
__global__ void context_kernel(const float* __restrict__ w, const __bf16* __restrict__ enc,
                               float* __restrict__ ctx) {
  int j = blockIdx.x * 256 + threadIdx.x;   // 0..1023
  int b = blockIdx.y;
  const __bf16* ep = enc + (size_t)b * 512 * 1024 + j;
  const float* wb = w + b * 512;
  float s = 0.f;
  for (int sidx = 0; sidx < 512; ++sidx) s += wb[sidx] * bf2f(ep[(size_t)sidx * 1024]);
  ctx[b * 1024 + j] = s;
}

// A_dec[b] = [ e(256) | context(1024) | h(512) ] in bf16; also keep e in f32
__global__ void pack_dec_kernel(const float* __restrict__ trg, const float* __restrict__ dpW,
                                const float* __restrict__ dpb, const float* __restrict__ ctx,
                                const __bf16* __restrict__ hbf, __bf16* __restrict__ A,
                                float* __restrict__ ef, int t) {
  int col = blockIdx.x * 256 + threadIdx.x;  // 0..1791
  int b = blockIdx.y;
  if (col < 256) {
    float e = trg[b * 96 + t] * (1.0f / 1000.0f) * dpW[col] + dpb[col];
    ef[b * 256 + col] = e;
    A[b * 1792 + col] = f2bf(e);
  } else if (col < 1280) {
    A[b * 1792 + col] = f2bf(ctx[b * 1024 + (col - 256)]);
  } else {
    A[b * 1792 + col] = hbf[b * 512 + (col - 1280)];
  }
}

__global__ void dec_point_kernel(const float* __restrict__ gates, float* __restrict__ h,
                                 float* __restrict__ c, __bf16* __restrict__ hbf) {
  int idx = blockIdx.x * 256 + threadIdx.x;  // 64*512
  int j = idx & 511, b = idx >> 9;
  const float* g = gates + (size_t)b * 2048;
  float gi = g[j], gf = g[512 + j], gg = g[1024 + j], go = g[1536 + j];
  float cn = sigmoidf(gf) * c[b * 512 + j] + sigmoidf(gi) * tanhf(gg);
  float hn = sigmoidf(go) * tanhf(cn);
  c[b * 512 + j] = cn; h[b * 512 + j] = hn; hbf[b * 512 + j] = f2bf(hn);
}

// feat = [h_new(512) | ctx(1024) | e(256)]; pv = (feat.val_W + vb)*1000 ; sl = feat.stop_W + sb
__global__ void val_stop_kernel(const float* __restrict__ hdec, const float* __restrict__ ctx,
                                const float* __restrict__ ef, const float* __restrict__ valW,
                                const float* __restrict__ valb, const float* __restrict__ stopW,
                                const float* __restrict__ stopb, float* __restrict__ out, int t) {
  int b = blockIdx.x, tid = threadIdx.x;   // 256 threads
  float sv = 0.f, ss = 0.f;
  for (int k = tid; k < 1792; k += 256) {
    float f = (k < 512) ? hdec[b * 512 + k]
            : (k < 1536) ? ctx[b * 1024 + (k - 512)]
                         : ef[b * 256 + (k - 1536)];
    sv += f * valW[k]; ss += f * stopW[k];
  }
#pragma unroll
  for (int o = 16; o; o >>= 1) { sv += __shfl_xor(sv, o, 32); ss += __shfl_xor(ss, o, 32); }
  __shared__ float shv[8], shs[8];
  if ((tid & 31) == 0) { shv[tid >> 5] = sv; shs[tid >> 5] = ss; }
  __syncthreads();
  if (tid == 0) {
    float a = 0.f, c2 = 0.f;
#pragma unroll
    for (int i = 0; i < 8; ++i) { a += shv[i]; c2 += shs[i]; }
    out[b * 95 + t] = (a + valb[0]) * 1000.0f;
    out[64 * 95 + b * 95 + t] = c2 + stopb[0];
  }
}

// ---------------- host orchestration ----------------
extern "C" void kernel_launch(void* const* d_in, const int* in_sizes, int n_in,
                              void* d_out, int out_size, void* d_ws, size_t ws_size,
                              hipStream_t stream) {
  (void)in_sizes; (void)n_in; (void)out_size; (void)ws_size;
  const float* src       = (const float*)d_in[0];
  const int*   src_mask  = (const int*)  d_in[1];
  const float* trg       = (const float*)d_in[2];
  const float* enc_proj_W = (const float*)d_in[4];
  const float* enc_proj_b = (const float*)d_in[5];
  const float* Wih_f = (const float*)d_in[6];
  const float* Whh_f = (const float*)d_in[7];
  const float* bih_f = (const float*)d_in[8];
  const float* bhh_f = (const float*)d_in[9];
  const float* Wih_b = (const float*)d_in[10];
  const float* Whh_b = (const float*)d_in[11];
  const float* bih_b = (const float*)d_in[12];
  const float* bhh_b = (const float*)d_in[13];
  const float* fc_h_W = (const float*)d_in[14];
  const float* fc_h_b = (const float*)d_in[15];
  const float* fc_c_W = (const float*)d_in[16];
  const float* fc_c_b = (const float*)d_in[17];
  const float* dec_proj_W = (const float*)d_in[18];
  const float* dec_proj_b = (const float*)d_in[19];
  const float* attn_Wenc = (const float*)d_in[20];
  const float* attn_Wdec = (const float*)d_in[21];
  const float* attn_v    = (const float*)d_in[22];
  const float* dec_Wih = (const float*)d_in[23];
  const float* dec_Whh = (const float*)d_in[24];
  const float* dec_bih = (const float*)d_in[25];
  const float* dec_bhh = (const float*)d_in[26];
  const float* val_W  = (const float*)d_in[27];
  const float* val_b  = (const float*)d_in[28];
  const float* stop_W = (const float*)d_in[29];
  const float* stop_b = (const float*)d_in[30];
  float* out = (float*)d_out;

  // workspace carve (all sizes multiples of 256B, ~170 MB total)
  char* ws = (char*)d_ws; size_t off = 0;
  auto alloc = [&](size_t bytes) -> char* {
    char* p = ws + off; off = (off + bytes + 255) & ~(size_t)255; return p;
  };
  __bf16* emb    = (__bf16*)alloc((size_t)512 * 64 * 256 * 2);     // [S][B][E]
  __bf16* WcatF  = (__bf16*)alloc((size_t)2048 * 768 * 2);          // [Wih_f|Whh_f]
  __bf16* WcatB  = (__bf16*)alloc((size_t)2048 * 768 * 2);
  __bf16* WencB  = (__bf16*)alloc((size_t)512 * 1024 * 2);
  __bf16* WdecB  = (__bf16*)alloc((size_t)512 * 512 * 2);
  __bf16* Wdcat  = (__bf16*)alloc((size_t)2048 * 1792 * 2);         // [dec_Wih|dec_Whh]
  float* biasF = (float*)alloc(2048 * 4);
  float* biasB = (float*)alloc(2048 * 4);
  float* biasD = (float*)alloc(2048 * 4);
  float*  h_enc    = (float*)alloc((size_t)2 * 64 * 512 * 4);       // zeroed region start
  float*  c_enc    = (float*)alloc((size_t)2 * 64 * 512 * 4);
  __bf16* h_enc_bf = (__bf16*)alloc((size_t)2 * 64 * 512 * 2);      // zeroed region end
  float*  gates_enc = (float*)alloc((size_t)2 * 64 * 2048 * 4);
  __bf16* enc_out_bf = (__bf16*)alloc((size_t)64 * 512 * 1024 * 2); // [B][S][2H]
  float*  eh = (float*)alloc((size_t)64 * 512 * 512 * 4);           // [B][S][H]
  float*  h_dec = (float*)alloc((size_t)64 * 512 * 4);
  float*  c_dec = (float*)alloc((size_t)64 * 512 * 4);
  __bf16* h_dec_bf = (__bf16*)alloc((size_t)64 * 512 * 2);
  float* hp     = (float*)alloc((size_t)64 * 512 * 4);
  float* scores = (float*)alloc((size_t)64 * 512 * 4);
  float* wsm    = (float*)alloc((size_t)64 * 512 * 4);
  float* ctx    = (float*)alloc((size_t)64 * 1024 * 4);
  float* ef     = (float*)alloc((size_t)64 * 256 * 4);
  __bf16* A_dec = (__bf16*)alloc((size_t)64 * 1792 * 2);
  float* gates_dec = (float*)alloc((size_t)64 * 2048 * 4);

  // ---- prep: weight conversion / concatenation, biases, embedding, state zeroing ----
  catcvt_kernel<<<dim3(2048 * 768 / 256), 256, 0, stream>>>(WcatF, Wih_f, 256, Whh_f, 512, 2048);
  catcvt_kernel<<<dim3(2048 * 768 / 256), 256, 0, stream>>>(WcatB, Wih_b, 256, Whh_b, 512, 2048);
  catcvt_kernel<<<dim3(2048 * 1792 / 256), 256, 0, stream>>>(Wdcat, dec_Wih, 1280, dec_Whh, 512, 2048);
  cvt_kernel<<<dim3(512 * 1024 / 256), 256, 0, stream>>>(WencB, attn_Wenc, 512 * 1024);
  cvt_kernel<<<dim3(512 * 512 / 256), 256, 0, stream>>>(WdecB, attn_Wdec, 512 * 512);
  addvec_kernel<<<dim3(8), 256, 0, stream>>>(biasF, bih_f, bhh_f, 2048);
  addvec_kernel<<<dim3(8), 256, 0, stream>>>(biasB, bih_b, bhh_b, 2048);
  addvec_kernel<<<dim3(8), 256, 0, stream>>>(biasD, dec_bih, dec_bhh, 2048);
  embed_kernel<<<dim3(512 * 64 * 256 / 256), 256, 0, stream>>>(src, enc_proj_W, enc_proj_b, emb);
  zero_kernel<<<dim3(640), 256, 0, stream>>>((unsigned*)h_enc, 163840); // h_enc+c_enc+h_enc_bf

  // ---- encoder: 512 serial steps, both directions per launch ----
  for (int t = 0; t < 512; ++t) {
    enc_step_gemm_kernel<<<dim3(8, 2, 2), 256, 0, stream>>>(
        emb, h_enc_bf, WcatF, WcatB, biasF, biasB, gates_enc, t);
    enc_step_point_kernel<<<dim3(256), 256, 0, stream>>>(
        gates_enc, src_mask, h_enc, c_enc, h_enc_bf, enc_out_bf, t);
  }

  // ---- bridge: hidden/cell init ----
  fc_init_kernel<<<dim3(256), 256, 0, stream>>>(h_enc, c_enc, fc_h_W, fc_h_b,
                                                fc_c_W, fc_c_b, h_dec, c_dec, h_dec_bf);

  // ---- hoisted attention projection: eh = enc_outputs @ Wenc^T  (M=32768,N=512,K=1024) ----
  gemm_bf16_kernel<<<dim3(2, 1024), 256, 0, stream>>>(enc_out_bf, 1024, WencB, 1024,
                                                      eh, 512, 1024, (const float*)nullptr);

  // ---- decoder: 95 serial steps ----
  for (int t = 0; t < 95; ++t) {
    gemm_bf16_kernel<<<dim3(2, 2), 256, 0, stream>>>(h_dec_bf, 512, WdecB, 512,
                                                     hp, 512, 512, (const float*)nullptr);
    attn_score_kernel<<<dim3(64, 64), 256, 0, stream>>>(hp, eh, attn_v, src_mask, scores);
    softmax512_kernel<<<dim3(64), 256, 0, stream>>>(scores, wsm);
    context_kernel<<<dim3(4, 64), 256, 0, stream>>>(wsm, enc_out_bf, ctx);
    pack_dec_kernel<<<dim3(7, 64), 256, 0, stream>>>(trg, dec_proj_W, dec_proj_b,
                                                     ctx, h_dec_bf, A_dec, ef, t);
    gemm_bf16_kernel<<<dim3(8, 2), 256, 0, stream>>>(A_dec, 1792, Wdcat, 1792,
                                                     gates_dec, 2048, 1792, biasD);
    dec_point_kernel<<<dim3(128), 256, 0, stream>>>(gates_dec, h_dec, c_dec, h_dec_bf);
    val_stop_kernel<<<dim3(64), 256, 0, stream>>>(h_dec, ctx, ef, val_W, val_b,
                                                  stop_W, stop_b, out, t);
  }
}
